// decoder_self_block_10161892623138
// MI455X (gfx1250) — matrix operand order repaired and checked by reading, not compiled, not run on hardware
//
#include <hip/hip_runtime.h>
#include <hip/hip_bf16.h>

typedef _Float16 half_t;
typedef __attribute__((ext_vector_type(16))) _Float16 v16h;
typedef __attribute__((ext_vector_type(8)))  _Float16 v8h;
typedef __attribute__((ext_vector_type(8)))  float    v8f;

#define B_   8
#define N_   4096
#define M_   2048
#define DIN  256
#define DOUT 128

__device__ __forceinline__ v8f wmma_f16(v16h a, v16h b, v8f c) {
    return __builtin_amdgcn_wmma_f32_16x16x32_f16(false, a, false, b, (short)0, c, false, false);
}

// Build A-matrix lane fragment from two contiguous 8-half runs (16B aligned).
__device__ __forceinline__ v16h load_a_frag(const half_t* p) {
    v8h lo = *(const v8h*)p;
    v8h hi = *(const v8h*)(p + 8);
    return __builtin_shufflevector(lo, hi, 0,1,2,3,4,5,6,7,8,9,10,11,12,13,14,15);
}

// ---------------- TDM: 1-D contiguous f16 copy global -> LDS ----------------
#if __has_builtin(__builtin_amdgcn_tensor_load_to_lds)
#define HAVE_TDM 1
typedef unsigned int u32x4 __attribute__((ext_vector_type(4)));
typedef int          i32x8 __attribute__((ext_vector_type(8)));
typedef int          i32x4 __attribute__((ext_vector_type(4)));

__device__ __forceinline__ void tdm_copy_f16_1d(void* lds_dst, const void* gsrc,
                                                unsigned nelem /* f16 elems, <= 65535 */) {
    unsigned long long ga = (unsigned long long)(uintptr_t)gsrc;
    unsigned la = (unsigned)(uintptr_t)lds_dst;        // LDS byte offset (addr[31:0])
    u32x4 g0;
    g0[0] = 1u;                                        // count=1, user descriptor
    g0[1] = la;                                        // lds_addr
    g0[2] = (unsigned)ga;                              // global_addr[31:0]
    g0[3] = ((unsigned)(ga >> 32) & 0x01FFFFFFu)       // global_addr[56:32]
            | (2u << 30);                              // type = 2 ("image")
    i32x8 g1;
    g1[0] = 0x00010000;                                // data_size=1 (2 bytes)
    g1[1] = (int)((nelem & 0xFFFFu) << 16);            // tensor_dim0[15:0] @ bits 63:48
    g1[2] = (int)(((nelem >> 16) & 0xFFFFu)            // tensor_dim0[31:16]
                  | (1u << 16));                       // tensor_dim1 = 1
    g1[3] = (int)((nelem & 0xFFFFu) << 16);            // tile_dim0 @ bits 127:112
    g1[4] = 1;                                         // tile_dim1 = 1
    g1[5] = (int)nelem;                                // tensor_dim0_stride
    g1[6] = 0;
    g1[7] = 0;
    i32x4 z4 = {0, 0, 0, 0};
    i32x8 z8 = {0, 0, 0, 0, 0, 0, 0, 0};
    // 6-arg toolchain signature: (g0, g1, g2, g3, extra, cpol)
    __builtin_amdgcn_tensor_load_to_lds(g0, g1, z4, z4, z8, 0);
}

__device__ __forceinline__ void tdm_wait0() {
#if __has_builtin(__builtin_amdgcn_s_wait_tensorcnt)
    __builtin_amdgcn_s_wait_tensorcnt(0);
#else
    asm volatile("s_wait_tensorcnt 0x0" ::: "memory");
#endif
}
#endif

// ---------------- kernel 0: weights -> f16, transposed [DOUT][DIN] ----------------
__global__ __launch_bounds__(256) void k_prep(const float* __restrict__ wq,
                                              const float* __restrict__ wk,
                                              const float* __restrict__ wv,
                                              half_t* __restrict__ wtq,
                                              half_t* __restrict__ wtk,
                                              half_t* __restrict__ wtv) {
    int i = blockIdx.x * 256 + threadIdx.x;
    if (i >= DIN * DOUT) return;
    int k = i / DOUT, n = i % DOUT;
    wtq[n * DIN + k] = (half_t)wq[i];
    wtk[n * DIN + k] = (half_t)wk[i];
    wtv[n * DIN + k] = (half_t)wv[i];
}

// ---------------- kernel 1: gather + Q/K/V projection (16 rows / block) ----------------
__global__ __launch_bounds__(128) void k_qkv(const float* __restrict__ Y,
                                             const int* __restrict__ idx,
                                             const half_t* __restrict__ wtq,
                                             const half_t* __restrict__ wtk,
                                             const half_t* __restrict__ wtv,
                                             const float* __restrict__ bq,
                                             const float* __restrict__ bk,
                                             const float* __restrict__ bv,
                                             half_t* __restrict__ Qh,
                                             half_t* __restrict__ Kh,
                                             half_t* __restrict__ VhT,
                                             float*  __restrict__ Vf) {
    __shared__ alignas(16) half_t g_lds[16 * DIN];  // 8 KB, gathered rows as f16
    __shared__ int s_idx[16];

    const int blk = blockIdx.x;
    const int b  = blk >> 7;          // 128 m-tiles per batch
    const int m0 = (blk & 127) * 16;
    const int tid = threadIdx.x;

    if (tid < 16) s_idx[tid] = idx[b * M_ + m0 + tid];
    __syncthreads();

    {   // cooperative gather: thread -> row tid>>3, cols (tid&7)*32..+31
        int r = tid >> 3, c0 = (tid & 7) * 32;
        const float* src = Y + ((size_t)b * N_ + s_idx[r]) * DIN + c0;
        #pragma unroll
        for (int j = 0; j < 32; j++) g_lds[r * DIN + c0 + j] = (half_t)src[j];
    }
    __syncthreads();

    const int wave = tid >> 5, lane = tid & 31;
    const int ln = lane & 15, hi = lane >> 4;
    const int n0 = wave * 32;  // each wave owns 2 column tiles of 16

    v8f accq[2] = {{}, {}}, acck[2] = {{}, {}}, accv[2] = {{}, {}};

    for (int kk = 0; kk < 8; kk++) {           // K = 256 in steps of 32
        const int k0 = kk * 32;
        v16h a = load_a_frag(&g_lds[ln * DIN + k0 + hi * 16]);
        const int kb = k0 + hi * 16;
        #pragma unroll
        for (int t = 0; t < 2; t++) {
            const int n = n0 + t * 16 + ln;    // B lane column
            v16h bqv = *(const v16h*)(wtq + n * DIN + kb);
            v16h bkv = *(const v16h*)(wtk + n * DIN + kb);
            v16h bvv = *(const v16h*)(wtv + n * DIN + kb);
            accq[t] = wmma_f16(a, bqv, accq[t]);
            acck[t] = wmma_f16(a, bkv, acck[t]);
            accv[t] = wmma_f16(a, bvv, accv[t]);
        }
    }

    #pragma unroll
    for (int t = 0; t < 2; t++) {
        const int col = n0 + t * 16 + ln;
        const float biq = bq[col], bik = bk[col], biv = bv[col];
        #pragma unroll
        for (int r = 0; r < 8; r++) {
            const int m = m0 + r + hi * 8;      // D layout: vgpr r -> row
            const size_t o = ((size_t)b * M_ + m) * DOUT + col;
            float q = accq[t][r] + biq;
            float k = acck[t][r] + bik;
            float v = accv[t][r] + biv;
            v = v > 0.f ? v : 0.f;              // ReLU
            Qh[o] = (half_t)q;
            Kh[o] = (half_t)k;
            Vf[o] = v;
            VhT[((size_t)b * DOUT + col) * M_ + m] = (half_t)v;  // transposed for att@V
        }
    }
}

// ---------------- kernel 2: attention for one 16-query tile ----------------
__global__ __launch_bounds__(128) void k_attn(const half_t* __restrict__ Qh,
                                              const half_t* __restrict__ Kh,
                                              const half_t* __restrict__ VhT,
                                              const float*  __restrict__ Vf,
                                              float* __restrict__ out,
                                              float* __restrict__ att) {
    __shared__ alignas(16) float  s_scores[16 * M_];  // 128 KB: full score row-block
    __shared__ alignas(16) half_t s_atth[16 * M_];    //  64 KB: normalized att as f16
    __shared__ alignas(16) half_t s_q[16 * DOUT];     //   4 KB
    __shared__ float s_red[16 * 8];

    const int blk = blockIdx.x;
    const int b  = blk >> 7;
    const int m0 = (blk & 127) * 16;
    const int tid = threadIdx.x;
    const int wave = tid >> 5, lane = tid & 31;
    const int ln = lane & 15, hi = lane >> 4;

    // ---- stage Q tile (16x128 f16, contiguous 4KB) into LDS via TDM ----
#if defined(HAVE_TDM)
    if (tid < 32) {                       // wave 0 issues one DMA; others skip
        tdm_copy_f16_1d(s_q, Qh + ((size_t)b * M_ + m0) * DOUT, 16 * DOUT);
        tdm_wait0();
    }
#else
    for (int j = tid; j < 16 * DOUT; j += 128)
        s_q[j] = Qh[((size_t)b * M_ + m0 + (j >> 7)) * DOUT + (j & 127)];
#endif
    __syncthreads();

    // ---- phase 1: scores = Q . K^T / 16 ----
    const float scale = 0.0625f;
    #pragma unroll 2
    for (int nt = wave * 32; nt < wave * 32 + 32; nt++) {
        const int key0 = nt * 16;
        // prefetch next key tile's column for this lane
        __builtin_prefetch(Kh + ((size_t)b * M_ + key0 + 16 + ln) * DOUT, 0, 0);
        v8f acc = {};
        #pragma unroll
        for (int kc = 0; kc < 4; kc++) {      // K = 128 in steps of 32
            const int k0 = kc * 32;
            v16h a = load_a_frag(&s_q[ln * DOUT + k0 + hi * 16]);
            const int key = key0 + ln;        // B lane column = key
            v16h bv = *(const v16h*)(Kh + ((size_t)b * M_ + key) * DOUT + k0 + hi * 16);
            acc = wmma_f16(a, bv, acc);
        }
        #pragma unroll
        for (int r = 0; r < 8; r++)
            s_scores[(r + hi * 8) * M_ + key0 + ln] = acc[r] * scale;
    }
    __syncthreads();

    // ---- phase 2: softmax per row (16 rows x 2048), 8 threads per row ----
    const int srow = tid >> 3, sch = tid & 7;
    float* rowp = &s_scores[srow * M_];
    float mx = -3.402823466e38f;
    for (int c = sch * 256; c < sch * 256 + 256; c++) mx = fmaxf(mx, rowp[c]);
    s_red[srow * 8 + sch] = mx;
    __syncthreads();
    mx = s_red[srow * 8];
    #pragma unroll
    for (int j = 1; j < 8; j++) mx = fmaxf(mx, s_red[srow * 8 + j]);
    __syncthreads();
    float sum = 0.f;
    for (int c = sch * 256; c < sch * 256 + 256; c++) {
        float e = __expf(rowp[c] - mx);
        rowp[c] = e;
        sum += e;
    }
    s_red[srow * 8 + sch] = sum;
    __syncthreads();
    sum = 0.f;
    #pragma unroll
    for (int j = 0; j < 8; j++) sum += s_red[srow * 8 + j];
    const float inv = 1.0f / sum;
    float* attg = att + ((size_t)b * M_ + m0 + srow) * M_;
    for (int c = sch * 256; c < sch * 256 + 256; c++) {
        float a = rowp[c] * inv;
        attg[c] = a;                          // output: normalized attention
        s_atth[srow * M_ + c] = (half_t)a;    // f16 copy for att@V
    }
    __syncthreads();

    // ---- phase 3: att_emb = att . V  (A from LDS f16, B from V^T) ----
    v8f emb[2] = {{}, {}};
    #pragma unroll 2
    for (int kt = 0; kt < 64; kt++) {         // K = 2048 keys in steps of 32
        const int k0 = kt * 32;
        __builtin_prefetch(VhT + ((size_t)b * DOUT + wave * 32 + ln) * M_ + k0 + 32, 0, 0);
        v16h a = load_a_frag(&s_atth[ln * M_ + k0 + hi * 16]);
        #pragma unroll
        for (int t = 0; t < 2; t++) {
            const int col = wave * 32 + t * 16 + ln;
            v16h bv = *(const v16h*)(VhT + ((size_t)b * DOUT + col) * M_ + k0 + hi * 16);
            emb[t] = wmma_f16(a, bv, emb[t]);
        }
    }
    float* s_emb = s_scores;                  // reuse scores LDS as [16][128] f32
    #pragma unroll
    for (int t = 0; t < 2; t++) {
        const int col = wave * 32 + t * 16 + ln;
        #pragma unroll
        for (int r = 0; r < 8; r++) s_emb[(r + hi * 8) * DOUT + col] = emb[t][r];
    }
    __syncthreads();

    // ---- phase 4: out = l2norm(V + l2norm(att_emb)) ----
    const int c0 = sch * 16;
    float ss = 0.f;
    #pragma unroll
    for (int j = 0; j < 16; j++) { float e = s_emb[srow * DOUT + c0 + j]; ss += e * e; }
    s_red[srow * 8 + sch] = ss;
    __syncthreads();
    ss = 0.f;
    #pragma unroll
    for (int j = 0; j < 8; j++) ss += s_red[srow * 8 + j];
    const float rs1 = rsqrtf(fmaxf(ss, 1e-12f));

    const float* vrow = Vf + ((size_t)b * M_ + m0 + srow) * DOUT;
    float tmp[16];
    float ss2 = 0.f;
    #pragma unroll
    for (int j = 0; j < 16; j++) {
        float t = vrow[c0 + j] + s_emb[srow * DOUT + c0 + j] * rs1;
        tmp[j] = t;
        ss2 += t * t;
    }
    __syncthreads();
    s_red[srow * 8 + sch] = ss2;
    __syncthreads();
    ss2 = 0.f;
    #pragma unroll
    for (int j = 0; j < 8; j++) ss2 += s_red[srow * 8 + j];
    const float rs2 = rsqrtf(fmaxf(ss2, 1e-12f));

    float* outg = out + ((size_t)b * M_ + m0 + srow) * DOUT;
    #pragma unroll
    for (int j = 0; j < 16; j++) outg[c0 + j] = tmp[j] * rs2;
}

extern "C" void kernel_launch(void* const* d_in, const int* in_sizes, int n_in,
                              void* d_out, int out_size, void* d_ws, size_t ws_size,
                              hipStream_t stream) {
    (void)in_sizes; (void)n_in; (void)out_size; (void)ws_size;
    const float* Y   = (const float*)d_in[0];
    const int*   idx = (const int*)d_in[1];
    const float* wq  = (const float*)d_in[2];
    const float* bq  = (const float*)d_in[3];
    const float* wk  = (const float*)d_in[4];
    const float* bk  = (const float*)d_in[5];
    const float* wv  = (const float*)d_in[6];
    const float* bv  = (const float*)d_in[7];

    float* out = (float*)d_out;                       // [8,2048,128]
    float* att = out + (size_t)B_ * M_ * DOUT;        // [8,2048,2048]

    // workspace partition (f16 halves unless noted); all offsets 16B-aligned
    half_t* wtq = (half_t*)d_ws;
    half_t* wtk = wtq + (size_t)DOUT * DIN;
    half_t* wtv = wtk + (size_t)DOUT * DIN;
    half_t* Qh  = wtv + (size_t)DOUT * DIN;
    half_t* Kh  = Qh  + (size_t)B_ * M_ * DOUT;
    half_t* VhT = Kh  + (size_t)B_ * M_ * DOUT;
    float*  Vf  = (float*)(VhT + (size_t)B_ * M_ * DOUT);

    k_prep<<<(DIN * DOUT + 255) / 256, 256, 0, stream>>>(wq, wk, wv, wtq, wtk, wtv);
    k_qkv<<<B_ * (M_ / 16), 128, 0, stream>>>(Y, idx, wtq, wtk, wtv, bq, bk, bv,
                                              Qh, Kh, VhT, Vf);
    k_attn<<<B_ * (M_ / 16), 128, 0, stream>>>(Qh, Kh, VhT, Vf, out, att);
}
